// FlashGeneAttentionLayer_34995393528236
// MI455X (gfx1250) — matrix operand order repaired and checked by reading, not compiled, not run on hardware
//
#include <hip/hip_runtime.h>

// Problem constants (reference: B=2, S=2048, D=512, H=8, HD=64)
#define S_LEN 2048
#define D_MODEL 512
#define N_HEAD 8
#define HEAD_D 64
#define N_TOK 4096  // B * S

#ifndef __has_builtin
#define __has_builtin(x) 0
#endif

#if defined(__AMDGCN__) &&                                        \
    __has_builtin(__builtin_amdgcn_global_load_async_to_lds_b128) && \
    __has_builtin(__builtin_amdgcn_s_wait_asynccnt)
#define USE_ASYNC_LDS 1
#else
#define USE_ASYNC_LDS 0
#endif

typedef __attribute__((ext_vector_type(16))) __bf16 bf16x16;
typedef __attribute__((ext_vector_type(8))) float f32x8;

union BFrag {
  bf16x16 v;
  uint4 q[2];
};

__device__ __forceinline__ unsigned short f2bf(float f) {
  unsigned int u = __float_as_uint(f);
  u += 0x7FFFu + ((u >> 16) & 1u);  // round-to-nearest-even
  return (unsigned short)(u >> 16);
}

__device__ __forceinline__ f32x8 wmma_bf16(const BFrag& a, const BFrag& b, f32x8 c) {
  return __builtin_amdgcn_wmma_f32_16x16x32_bf16(false, a.v, false, b.v,
                                                 (short)0, c, false, false);
}

#if USE_ASYNC_LDS
// Builtin signature (from hipcc diagnostic): param 0 is
//   "__attribute__((__vector_size__(4*sizeof(int)))) int __device__ *"
// i.e. v4i32 in the global (AS1) language address space; dst is the LDS (AS3)
// counterpart.  Use the exact gcc-style vector type.
typedef int v4i __attribute__((vector_size(16)));
typedef __attribute__((address_space(1))) v4i* gv4p;
typedef __attribute__((address_space(3))) v4i* lv4p;
#endif

// Copy 64 bytes global -> LDS for this thread (async if available).
__device__ __forceinline__ void stage64B(const unsigned short* g, unsigned short* l) {
#if USE_ASYNC_LDS
  gv4p gp = (gv4p)(v4i*)g;  // reinterpret pointee, then generic->AS1
  lv4p lp = (lv4p)(v4i*)l;  // reinterpret pointee, then generic->AS3
  __builtin_amdgcn_global_load_async_to_lds_b128(gp + 0, lp + 0, 0, 0);
  __builtin_amdgcn_global_load_async_to_lds_b128(gp + 1, lp + 1, 0, 0);
  __builtin_amdgcn_global_load_async_to_lds_b128(gp + 2, lp + 2, 0, 0);
  __builtin_amdgcn_global_load_async_to_lds_b128(gp + 3, lp + 3, 0, 0);
#else
  const uint4* gq = (const uint4*)g;
  uint4* lq = (uint4*)l;
  lq[0] = gq[0];
  lq[1] = gq[1];
  lq[2] = gq[2];
  lq[3] = gq[3];
#endif
}

__device__ __forceinline__ void wait_async_lds() {
#if USE_ASYNC_LDS
  __builtin_amdgcn_s_wait_asynccnt(0);
#endif
}

// ---------------------------------------------------------------------------
// fp32 -> bf16 conversion
// ---------------------------------------------------------------------------
__global__ void cvt_bf16_kernel(const float* __restrict__ in,
                                unsigned short* __restrict__ out, int n) {
  int i = blockIdx.x * blockDim.x + threadIdx.x;
  if (i < n) out[i] = f2bf(in[i]);
}

// ---------------------------------------------------------------------------
// out = A @ W^T + bias.  A: [N_TOK x 512] bf16, W: [512 x 512] bf16.
// If outF != nullptr : write f32 [N_TOK x 512] (final projection).
// Else               : write bf16 in (B, H, S, HD) layout (Q/K/V projections).
// Block = 128 threads (4 waves); wave tile = 16 rows x 64 cols.
// Fragments are double-buffered so VMEM overlaps the matrix pipe.
// ---------------------------------------------------------------------------
__global__ __launch_bounds__(128) void gemm_xwT_kernel(
    const unsigned short* __restrict__ A, const unsigned short* __restrict__ W,
    const float* __restrict__ bias, unsigned short* __restrict__ outHead,
    float* __restrict__ outF) {
  const int lane = threadIdx.x & 31;
  const int wave = threadIdx.x >> 5;
  const int l15 = lane & 15;
  const bool lo = lane < 16;
  const int m0 = blockIdx.x * 64 + wave * 16;
  const int n0 = blockIdx.y * 64;

  f32x8 acc[4] = {};
  const int ka = lo ? 0 : 16;  // A-frag K run base  (contiguous, matches B)
  const int kb = lo ? 0 : 16;  // B-frag K run base  (contiguous, matches A)
  const unsigned short* arow = A + (m0 + l15) * D_MODEL;
  const unsigned short* wrow = W + (n0 + l15) * D_MODEL;

  BFrag a[2];
  BFrag b[2][4];
  auto load_frags = [&](int k0, int buf) {
    const unsigned short* ap = arow + k0 + ka;
    a[buf].q[0] = *(const uint4*)ap;
    a[buf].q[1] = *(const uint4*)(ap + 8);
#pragma unroll
    for (int t = 0; t < 4; ++t) {
      const unsigned short* bp = wrow + t * 16 * D_MODEL + k0 + kb;
      b[buf][t].q[0] = *(const uint4*)bp;
      b[buf][t].q[1] = *(const uint4*)(bp + 8);
    }
  };

  load_frags(0, 0);
#pragma unroll
  for (int k0 = 0; k0 < D_MODEL; k0 += 32) {
    const int cur = (k0 >> 5) & 1;
    if (k0 + 32 < D_MODEL) load_frags(k0 + 32, cur ^ 1);
#pragma unroll
    for (int t = 0; t < 4; ++t) acc[t] = wmma_bf16(a[cur], b[cur][t], acc[t]);
  }

  const int rowOff = lo ? 0 : 8;  // C/D layout: lanes 16-31 hold M = v + 8
#pragma unroll
  for (int t = 0; t < 4; ++t) {
#pragma unroll
    for (int v = 0; v < 8; ++v) {
      int r = m0 + v + rowOff;
      int c = n0 + t * 16 + l15;
      float val = acc[t][v] + bias[c];
      if (outF) {
        outF[r * D_MODEL + c] = val;
      } else {
        int bb = r >> 11;         // token row -> batch
        int s = r & (S_LEN - 1);  //           -> seq pos
        int h = c >> 6;           // col       -> head
        int hd = c & 63;          //           -> head dim
        outHead[(((bb * N_HEAD + h) * S_LEN) + s) * HEAD_D + hd] = f2bf(val);
      }
    }
  }
}

// ---------------------------------------------------------------------------
// Flash attention with masked re-normalization:
//   out[q] = (sum_k e_k * M_k * V_k) / (sum_k e_k * M_k + eps * sum_k e_k)
// with e_k = exp(s_k - m), s_k = scale*QK + (M_k==0 ? -1e9 : 0).
// Block = 128 threads (4 waves) -> 64 q rows; loop over 64-key chunks.
// K chunks are double-buffered in LDS and staged with async global->LDS
// loads (overlapping the current chunk's 16 WMMAs); V is staged transposed.
// ---------------------------------------------------------------------------
__global__ __launch_bounds__(128) void attn_kernel(
    const unsigned short* __restrict__ Q, const unsigned short* __restrict__ Km,
    const unsigned short* __restrict__ Vm, const float* __restrict__ M,
    unsigned short* __restrict__ outA) {
  __shared__ __align__(16) unsigned short Kl[2][64 * 72];  // K chunks, row-major
  __shared__ __align__(16) unsigned short Vt[64 * 72];     // V chunk, transposed
  __shared__ __align__(16) unsigned short Pl[4][16 * 72];  // per-wave P scratch

  const int lane = threadIdx.x & 31;
  const int wave = threadIdx.x >> 5;
  const int l15 = lane & 15;
  const bool lo = lane < 16;
  const int rowOff = lo ? 0 : 8;
  const int bh = blockIdx.y;
  const int b = bh >> 3;
  const int h = bh & 7;
  const int q0 = blockIdx.x * 64 + wave * 16;
  const float scale = 0.125f;  // HD^-0.5

  // Q fragments (A layout), K-dim = head dim, two 32-slices
  BFrag aq0, aq1;
  {
    const unsigned short* qp = Q + (bh * S_LEN + q0 + l15) * HEAD_D + (lo ? 0 : 16);
    aq0.q[0] = *(const uint4*)qp;
    aq0.q[1] = *(const uint4*)(qp + 8);
    aq1.q[0] = *(const uint4*)(qp + 32);
    aq1.q[1] = *(const uint4*)(qp + 40);
  }

  float mrow[8], lsum[8], zsum[8];
#pragma unroll
  for (int v = 0; v < 8; ++v) {
    mrow[v] = -1e30f;
    lsum[v] = 0.f;
    zsum[v] = 0.f;
  }
  f32x8 o[4] = {};

  const int ldRow = threadIdx.x >> 1;         // 0..63: key row in chunk
  const int ldHalf = (threadIdx.x & 1) * 32;  // 0 or 32: head-dim half
  const int kbo = lo ? 0 : 16;
  const int klo = lo ? 0 : 16;

  // Preload K chunk 0 into buffer 0.
  stage64B(Km + (bh * S_LEN + ldRow) * HEAD_D + ldHalf, &Kl[0][ldRow * 72 + ldHalf]);

  for (int kk = 0; kk < S_LEN; kk += 64) {
    const int cur = (kk >> 6) & 1;

    // ---- stage V chunk (transposed) into LDS ----
    {
      uint4 vv[4];
      const uint4* gv = (const uint4*)(Vm + (bh * S_LEN + kk + ldRow) * HEAD_D + ldHalf);
      vv[0] = gv[0];
      vv[1] = gv[1];
      vv[2] = gv[2];
      vv[3] = gv[3];
      const unsigned short* vs = (const unsigned short*)vv;
#pragma unroll
      for (int i = 0; i < 32; ++i) Vt[(ldHalf + i) * 72 + ldRow] = vs[i];
    }
    if (kk + 64 < S_LEN)
      __builtin_prefetch(Vm + (bh * S_LEN + kk + 64 + ldRow) * HEAD_D + ldHalf, 0, 1);

    wait_async_lds();  // K(kk) resident in Kl[cur]
    __syncthreads();   // all waves: Kl[cur] + Vt ready

    // Kick off async K(kk+64) into the other buffer; overlaps compute below.
    // Safe: Kl[cur^1] was last read before the previous iteration's barrier.
    if (kk + 64 < S_LEN)
      stage64B(Km + (bh * S_LEN + kk + 64 + ldRow) * HEAD_D + ldHalf,
               &Kl[cur ^ 1][ldRow * 72 + ldHalf]);

    // ---- scores: 16 q x 64 k (4 tiles, 2 WMMA each over HD=64) ----
    f32x8 sc[4];
#pragma unroll
    for (int t = 0; t < 4; ++t) {
      const unsigned short* kp = &Kl[cur][(t * 16 + l15) * 72 + kbo];
      BFrag b0, b1;
      b0.q[0] = *(const uint4*)kp;
      b0.q[1] = *(const uint4*)(kp + 8);
      b1.q[0] = *(const uint4*)(kp + 32);
      b1.q[1] = *(const uint4*)(kp + 40);
      f32x8 c = {};
      c = wmma_bf16(aq0, b0, c);
      c = wmma_bf16(aq1, b1, c);
      sc[t] = c;
    }

    // ---- mask + scale ----
    float mv[4][8];
#pragma unroll
    for (int t = 0; t < 4; ++t) {
#pragma unroll
      for (int v = 0; v < 8; ++v) {
        int r = q0 + v + rowOff;
        int c = kk + t * 16 + l15;
        float m = M[(b * S_LEN + r) * S_LEN + c];
        mv[t][v] = m;
        sc[t][v] = sc[t][v] * scale + (m == 0.f ? -1e9f : 0.f);
      }
    }

    // ---- online softmax: row max across 16 lanes, rescale state ----
#pragma unroll
    for (int v = 0; v < 8; ++v) {
      float mx = fmaxf(fmaxf(sc[0][v], sc[1][v]), fmaxf(sc[2][v], sc[3][v]));
#pragma unroll
      for (int off = 1; off < 16; off <<= 1) mx = fmaxf(mx, __shfl_xor(mx, off, 32));
      float mn = fmaxf(mrow[v], mx);
      float al = __expf(mrow[v] - mn);
      mrow[v] = mn;
      lsum[v] *= al;
      zsum[v] *= al;
#pragma unroll
      for (int t = 0; t < 4; ++t) o[t][v] *= al;
    }

    // ---- p = exp * mask; accumulate l/z; spill P to LDS (C layout) ----
#pragma unroll
    for (int t = 0; t < 4; ++t) {
#pragma unroll
      for (int v = 0; v < 8; ++v) {
        float e = __expf(sc[t][v] - mrow[v]);
        zsum[v] += e;
        float p = e * mv[t][v];
        lsum[v] += p;
        Pl[wave][(v + rowOff) * 72 + t * 16 + l15] = f2bf(p);
      }
    }

    // ---- O += P @ V : 2 key-halves x 4 hd-tiles ----
#pragma unroll
    for (int kh = 0; kh < 2; ++kh) {
      BFrag ap;
      const unsigned short* pp = &Pl[wave][l15 * 72 + kh * 32 + klo];
      ap.q[0] = *(const uint4*)pp;
      ap.q[1] = *(const uint4*)(pp + 8);
#pragma unroll
      for (int t = 0; t < 4; ++t) {
        BFrag bv;
        const unsigned short* vp = &Vt[(t * 16 + l15) * 72 + kh * 32 + kbo];
        bv.q[0] = *(const uint4*)vp;
        bv.q[1] = *(const uint4*)(vp + 8);
        o[t] = wmma_bf16(ap, bv, o[t]);
      }
    }
    __syncthreads();  // everyone done with Vt / Kl[cur]
  }

  // ---- final cross-lane sums, divide, write bf16 token-major ----
#pragma unroll
  for (int v = 0; v < 8; ++v) {
#pragma unroll
    for (int off = 1; off < 16; off <<= 1) {
      lsum[v] += __shfl_xor(lsum[v], off, 32);
      zsum[v] += __shfl_xor(zsum[v], off, 32);
    }
  }
#pragma unroll
  for (int t = 0; t < 4; ++t) {
#pragma unroll
    for (int v = 0; v < 8; ++v) {
      int r = q0 + v + rowOff;
      float denom = lsum[v] + 1e-8f * zsum[v];
      float val = o[t][v] / denom;
      outA[(b * S_LEN + r) * D_MODEL + h * HEAD_D + t * 16 + l15] = f2bf(val);
    }
  }
}

// ---------------------------------------------------------------------------
extern "C" void kernel_launch(void* const* d_in, const int* in_sizes, int n_in,
                              void* d_out, int out_size, void* d_ws, size_t ws_size,
                              hipStream_t stream) {
  (void)in_sizes; (void)n_in; (void)out_size; (void)ws_size;
  const float* x = (const float*)d_in[0];
  const float* M = (const float*)d_in[1];
  const float* Wq = (const float*)d_in[2];
  const float* bq = (const float*)d_in[3];
  const float* Wk = (const float*)d_in[4];
  const float* bk = (const float*)d_in[5];
  const float* Wv = (const float*)d_in[6];
  const float* bv = (const float*)d_in[7];
  const float* Wo = (const float*)d_in[8];
  const float* bo = (const float*)d_in[9];
  float* out = (float*)d_out;

  // Workspace carve-up (bf16 tensors), ~22 MB total
  unsigned short* xb = (unsigned short*)d_ws;          // [N_TOK x 512]
  unsigned short* wqb = xb + (size_t)N_TOK * D_MODEL;  // [512 x 512]
  unsigned short* wkb = wqb + (size_t)D_MODEL * D_MODEL;
  unsigned short* wvb = wkb + (size_t)D_MODEL * D_MODEL;
  unsigned short* wob = wvb + (size_t)D_MODEL * D_MODEL;
  unsigned short* Qb = wob + (size_t)D_MODEL * D_MODEL;  // (B,H,S,HD)
  unsigned short* Kb = Qb + (size_t)N_TOK * D_MODEL;
  unsigned short* Vb = Kb + (size_t)N_TOK * D_MODEL;
  unsigned short* Ab = Vb + (size_t)N_TOK * D_MODEL;  // attn out, token-major

  cvt_bf16_kernel<<<N_TOK * D_MODEL / 256, 256, 0, stream>>>(x, xb, N_TOK * D_MODEL);
  cvt_bf16_kernel<<<D_MODEL * D_MODEL / 256, 256, 0, stream>>>(Wq, wqb, D_MODEL * D_MODEL);
  cvt_bf16_kernel<<<D_MODEL * D_MODEL / 256, 256, 0, stream>>>(Wk, wkb, D_MODEL * D_MODEL);
  cvt_bf16_kernel<<<D_MODEL * D_MODEL / 256, 256, 0, stream>>>(Wv, wvb, D_MODEL * D_MODEL);
  cvt_bf16_kernel<<<D_MODEL * D_MODEL / 256, 256, 0, stream>>>(Wo, wob, D_MODEL * D_MODEL);

  dim3 gg(N_TOK / 64, D_MODEL / 64);
  gemm_xwT_kernel<<<gg, 128, 0, stream>>>(xb, wqb, bq, Qb, nullptr);
  gemm_xwT_kernel<<<gg, 128, 0, stream>>>(xb, wkb, bk, Kb, nullptr);
  gemm_xwT_kernel<<<gg, 128, 0, stream>>>(xb, wvb, bv, Vb, nullptr);

  attn_kernel<<<dim3(S_LEN / 64, 2 * N_HEAD), 128, 0, stream>>>(Qb, Kb, Vb, M, Ab);

  gemm_xwT_kernel<<<gg, 128, 0, stream>>>(Ab, wob, bo, nullptr, out);
}
